// Transformer_11201274708409
// MI455X (gfx1250) — matrix operand order repaired and checked by reading, not compiled, not run on hardware
//
#include <hip/hip_runtime.h>
#include <hip/hip_bf16.h>
#include <math.h>

// ---------------------------------------------------------------------------
// BERT-style encoder forward for gfx1250 (CDNA5): bf16 WMMA + TDM staging.
// ---------------------------------------------------------------------------

typedef __attribute__((ext_vector_type(16))) __bf16 v16bf;
typedef __attribute__((ext_vector_type(8)))  __bf16 bf16x8;
typedef __attribute__((ext_vector_type(8)))  float  v8f;
typedef __attribute__((ext_vector_type(4)))  float  f32x4;
typedef __attribute__((ext_vector_type(4)))  unsigned u32x4;
typedef __attribute__((ext_vector_type(8)))  int    i32x8;
typedef __attribute__((ext_vector_type(4)))  int    i32x4;

#define SEQ    512
#define DMODEL 768
#define NHEAD  12
#define DKH    64
#define QKVW   2304   // 3*DMODEL
#define NTOK   8192   // B*S

#if defined(__HIP_DEVICE_COMPILE__) && \
    __has_builtin(__builtin_amdgcn_tensor_load_to_lds) && \
    __has_builtin(__builtin_amdgcn_s_wait_tensorcnt)
#define HAVE_TDM 1
#else
#define HAVE_TDM 0
#endif

__device__ __forceinline__ v8f wmma_bf16(v16bf a, v16bf b, v8f c) {
  return __builtin_amdgcn_wmma_f32_16x16x32_bf16(false, a, false, b, (short)0, c,
                                                 false, false);
}

__device__ __forceinline__ v16bf cat8(bf16x8 lo, bf16x8 hi) {
  return __builtin_shufflevector(lo, hi, 0, 1, 2, 3, 4, 5, 6, 7,
                                 8, 9, 10, 11, 12, 13, 14, 15);
}

// --- Tensor Data Mover: 2D tile (tile_d0 elems x tile_d1 rows, bf16) --------
// D# packing per CDNA5 ISA ch.8 (group0: count/lds/global/type, group1: dims).
// 6-arg builtin form (clang-23 / therock-10.0 lane): groups 2,3 and the extra
// int32x8 operand carry higher-dim fields -> all zero for a plain 2D tile.
__device__ __forceinline__ void tdm_load_2d(unsigned lds_addr, const void* gptr,
                                            unsigned tensor_d0, unsigned tensor_d1,
                                            unsigned tile_d0, unsigned tile_d1,
                                            unsigned long long stride0) {
#if HAVE_TDM
  unsigned long long ga = (unsigned long long)gptr;
  u32x4 g0;
  g0[0] = 1u;                                            // count=1, user mode
  g0[1] = lds_addr;                                      // LDS byte address
  g0[2] = (unsigned)(ga & 0xFFFFFFFFu);                  // global_addr[31:0]
  g0[3] = (unsigned)((ga >> 32) & 0x1FFFFFFu) | (2u << 30);  // [56:32] | type=2
  i32x8 g1;
  g1[0] = (int)(1u << 16);                               // data_size=2B, no mcast
  g1[1] = (int)((tensor_d0 & 0xFFFFu) << 16);            // tensor_dim0[15:0]
  g1[2] = (int)((tensor_d0 >> 16) | ((tensor_d1 & 0xFFFFu) << 16));
  g1[3] = (int)((tensor_d1 >> 16) | (tile_d0 << 16));    // tile_dim0
  g1[4] = (int)tile_d1;                                  // tile_dim1; dim2=0
  g1[5] = (int)(stride0 & 0xFFFFFFFFull);                // dim0_stride[31:0]
  g1[6] = (int)((stride0 >> 32) & 0xFFFFull);            // dim0_stride[47:32]
  g1[7] = 0;
  i32x4 z4 = {0, 0, 0, 0};
  i32x8 z8 = {0, 0, 0, 0, 0, 0, 0, 0};
  __builtin_amdgcn_tensor_load_to_lds(g0, g1, z4, z4, z8, 0);
#else
  (void)lds_addr; (void)gptr; (void)tensor_d0; (void)tensor_d1;
  (void)tile_d0; (void)tile_d1; (void)stride0;
#endif
}

template <short N>
__device__ __forceinline__ void tdm_wait() {
#if HAVE_TDM
  __builtin_amdgcn_s_wait_tensorcnt(N);
#endif
}

// ---------------------------------------------------------------------------
// Embedding: h[t,:] = (pos[s] + tok[x] + seg[tt]) * sqrt(D)
// ---------------------------------------------------------------------------
__global__ __launch_bounds__(256) void emb_kernel(
    const int* __restrict__ x, const int* __restrict__ tt,
    const float* __restrict__ tok, const float* __restrict__ pos,
    const float* __restrict__ seg, float* __restrict__ h) {
  int t = blockIdx.x;
  int s = t & (SEQ - 1);
  int id = x[t];
  int sg = tt[t];
  const float sc = 27.712812921102035f;  // sqrt(768)
  size_t base = (size_t)t * DMODEL;
#pragma unroll
  for (int j = 0; j < 3; ++j) {
    int c = threadIdx.x + j * 256;
    h[base + c] = sc * (pos[(size_t)s * DMODEL + c] +
                        tok[(size_t)id * DMODEL + c] +
                        seg[(size_t)sg * DMODEL + c]);
  }
}

// ---------------------------------------------------------------------------
// LayerNorm (torch semantics: unbiased std, g*(x-mean)/(std+eps)+b) -> bf16
// ---------------------------------------------------------------------------
__global__ __launch_bounds__(256) void ln_kernel(
    const float* __restrict__ x, const float* __restrict__ g,
    const float* __restrict__ b, __bf16* __restrict__ out) {
  int t = blockIdx.x;
  const float* row = x + (size_t)t * DMODEL;
  int tid = threadIdx.x, wave = tid >> 5, lane = tid & 31;
  __shared__ float red[8];

  float v[3];
  float s = 0.f;
#pragma unroll
  for (int j = 0; j < 3; ++j) { v[j] = row[tid + j * 256]; s += v[j]; }
#pragma unroll
  for (int m = 16; m >= 1; m >>= 1) s += __shfl_xor(s, m, 32);
  if (lane == 0) red[wave] = s;
  __syncthreads();
  float tot = 0.f;
#pragma unroll
  for (int w = 0; w < 8; ++w) tot += red[w];
  float mean = tot * (1.0f / 768.0f);
  __syncthreads();

  float ss = 0.f;
#pragma unroll
  for (int j = 0; j < 3; ++j) { float d = v[j] - mean; ss += d * d; }
#pragma unroll
  for (int m = 16; m >= 1; m >>= 1) ss += __shfl_xor(ss, m, 32);
  if (lane == 0) red[wave] = ss;
  __syncthreads();
  float tot2 = 0.f;
#pragma unroll
  for (int w = 0; w < 8; ++w) tot2 += red[w];
  float inv = 1.0f / (sqrtf(tot2 * (1.0f / 767.0f)) + 1e-6f);

  __bf16* orow = out + (size_t)t * DMODEL;
#pragma unroll
  for (int j = 0; j < 3; ++j) {
    int c = tid + j * 256;
    orow[c] = (__bf16)(g[c] * (v[j] - mean) * inv + b[c]);
  }
}

// ---------------------------------------------------------------------------
// Generic bf16 WMMA GEMM: C[M,N] = act(A[M,K] * W[N,K]^T + bias) (+ residual)
// Block tile 64(M) x 128(N), BK=32. 8 waves (2x4), each wave 32x32 (2x2 WMMA).
// FULL=true (exact-tile case): A tile staged by the Tensor Data Mover into a
// double-buffered LDS tile (DMA for k+1 overlaps current WMMAs); W staged via
// register prefetch with on-the-fly fp32->bf16 conversion.
// ---------------------------------------------------------------------------
#define GBM 64
#define GBN 128
#define GBK 32

template <bool FULL>
__global__ __launch_bounds__(256) void gemm_bf16_kernel(
    const __bf16* __restrict__ A, const float* __restrict__ W,
    const float* __restrict__ bias, const float* __restrict__ resid,
    float* __restrict__ outF, __bf16* __restrict__ outB,
    int M, int N, int K, int act) {
  constexpr bool USE_TDM = FULL && (HAVE_TDM != 0);
  __shared__ __bf16 As[2][GBM][GBK];   // double buffer (TDM path uses both)
  __shared__ __bf16 Bst[GBN][GBK];     // Bst[n][k] = W[n0+n][k0+k]

  int tid = threadIdx.x;
  int wave = tid >> 5, lane = tid & 31;
  int hl = lane >> 4, lcol = lane & 15;
  int m0 = blockIdx.y * GBM, n0 = blockIdx.x * GBN;
  int waveM = (wave >> 2) * 32;
  int waveN = (wave & 3) * 32;

  // staging assignments (per thread)
  int arow = tid >> 2, ac8 = (tid & 3) * 8;     // A: 8 bf16 per thread
  int bn = tid >> 1, bk16 = (tid & 1) * 16;     // W: 16 f32 per thread
  int gm = m0 + arow;
  int gn = n0 + bn;
  const __bf16* aptr = A + (size_t)gm * K + ac8;
  const float*  wptr = W + (size_t)gn * K + bk16;

  bf16x8 areg, breg0, breg1;
  auto load_a = [&](int k0) {
    if (FULL || gm < M) {
      areg = *(const bf16x8*)(aptr + k0);
    } else {
#pragma unroll
      for (int j = 0; j < 8; ++j) areg[j] = (__bf16)0.0f;
    }
  };
  auto load_w = [&](int k0) {
    if (FULL || gn < N) {
      const f32x4* wp = (const f32x4*)(wptr + k0);
      f32x4 w0 = wp[0], w1 = wp[1], w2 = wp[2], w3 = wp[3];
#pragma unroll
      for (int j = 0; j < 4; ++j) {
        breg0[j]     = (__bf16)w0[j];
        breg0[4 + j] = (__bf16)w1[j];
        breg1[j]     = (__bf16)w2[j];
        breg1[4 + j] = (__bf16)w3[j];
      }
    } else {
#pragma unroll
      for (int j = 0; j < 8; ++j) { breg0[j] = (__bf16)0.0f; breg1[j] = (__bf16)0.0f; }
    }
  };
  auto tdm_issue = [&](int nb, int k0) {
    tdm_load_2d((unsigned)(unsigned long long)&As[nb][0][0],
                A + (size_t)m0 * K + k0,
                (unsigned)K, (unsigned)M, GBK, GBM, (unsigned long long)K);
  };

  v8f acc[2][2];
#pragma unroll
  for (int i = 0; i < 2; ++i)
#pragma unroll
    for (int j = 0; j < 2; ++j)
#pragma unroll
      for (int r = 0; r < 8; ++r) acc[i][j][r] = 0.f;

  if constexpr (USE_TDM) {
    if (wave == 0) tdm_issue(0, 0);
  } else {
    load_a(0);
  }
  load_w(0);

  int p = 0;
  for (int k0 = 0; k0 < K; k0 += GBK) {
    __syncthreads();                       // previous compute done reading LDS
    *(bf16x8*)&Bst[bn][bk16] = breg0;
    *(bf16x8*)&Bst[bn][bk16 + 8] = breg1;
    if constexpr (!USE_TDM) *(bf16x8*)&As[0][arow][ac8] = areg;

    bool more = (k0 + GBK) < K;
    if (more) {
      if constexpr (USE_TDM) {
        if (wave == 0) tdm_issue(p ^ 1, k0 + GBK);   // DMA next tile
      } else {
        load_a(k0 + GBK);
      }
      load_w(k0 + GBK);                    // register prefetch (overlaps WMMA)
    }
    if constexpr (USE_TDM) {
      if (wave == 0) {                     // retire the older transfer only
        if (more) tdm_wait<1>(); else tdm_wait<0>();
      }
    }
    __syncthreads();

    __bf16(*Ab)[GBK] = As[USE_TDM ? p : 0];
    v16bf af[2], bv[2];
#pragma unroll
    for (int s = 0; s < 2; ++s) {
      const __bf16* rp = &Ab[waveM + s * 16 + lcol][0];
      af[s] = cat8(*(const bf16x8*)(rp + hl * 8),
                   *(const bf16x8*)(rp + 16 + hl * 8));
      bv[s] = cat8(*(const bf16x8*)&Bst[waveN + s * 16 + lcol][hl * 8],
                   *(const bf16x8*)&Bst[waveN + s * 16 + lcol][16 + hl * 8]);
    }
#pragma unroll
    for (int sm = 0; sm < 2; ++sm)
#pragma unroll
      for (int sn = 0; sn < 2; ++sn)
        acc[sm][sn] = wmma_bf16(af[sm], bv[sn], acc[sm][sn]);
    p ^= 1;
  }

  // Epilogue: C layout = VGPR r -> row (r + 8*hl), col = lcol
#pragma unroll
  for (int sm = 0; sm < 2; ++sm)
#pragma unroll
    for (int sn = 0; sn < 2; ++sn)
#pragma unroll
      for (int r = 0; r < 8; ++r) {
        int row = m0 + waveM + sm * 16 + r + hl * 8;
        int col = n0 + waveN + sn * 16 + lcol;
        if (FULL || (row < M && col < N)) {
          float v = acc[sm][sn][r];
          if (bias) v += bias[col];
          if (act == 1)      v = 0.5f * v * (1.0f + erff(v * 0.70710678118654752f));
          else if (act == 2) v = tanhf(v);
          else if (act == 3) v = fmaxf(v, 0.0f);
          if (resid) v += resid[(size_t)row * N + col];
          if (outF) outF[(size_t)row * N + col] = v;
          if (outB) outB[(size_t)row * N + col] = (__bf16)v;
        }
      }
}

// ---------------------------------------------------------------------------
// Flash attention: block = (b, h, 128 q rows), 8 waves x 16 q rows each.
// qkv: bf16 [8192, 2304] with cols [t*768 + h*64 + dk], t in {q,k,v}.
// obuf: bf16 [8192, 768] with cols [h*64 + dk].
// ---------------------------------------------------------------------------
__global__ __launch_bounds__(256) void attn_kernel(
    const __bf16* __restrict__ qkv, __bf16* __restrict__ obuf) {
  __shared__ __bf16 Ks[32][64];        // K block, row-major over dk
  __shared__ __bf16 Vt[64][32];        // V block transposed: Vt[dk][k]
  __shared__ __bf16 Pb[8][16][32];     // per-wave P tile (C->A relayout)

  int tid = threadIdx.x, wave = tid >> 5, lane = tid & 31;
  int hl = lane >> 4, lcol = lane & 15;
  int qblk = blockIdx.x * 128;
  int hh = blockIdx.y, bb = blockIdx.z;

  // Q fragments for this wave's 16 rows; fold in 1/sqrt(64) = 0.125
  int qrow = qblk + wave * 16 + lcol;
  const __bf16* qp = qkv + ((size_t)(bb * SEQ + qrow)) * QKVW + hh * DKH;
  v16bf a0 = cat8(*(const bf16x8*)(qp + hl * 8),
                  *(const bf16x8*)(qp + 16 + hl * 8));
  v16bf a1 = cat8(*(const bf16x8*)(qp + 32 + hl * 8),
                  *(const bf16x8*)(qp + 48 + hl * 8));
#pragma unroll
  for (int i = 0; i < 16; ++i) {
    a0[i] = (__bf16)((float)a0[i] * 0.125f);
    a1[i] = (__bf16)((float)a1[i] * 0.125f);
  }

  // K/V staging assignment (8 bf16 per thread each) + register prefetch
  int kr = tid >> 3, d8 = (tid & 7) * 8;
  const __bf16* kvb = qkv + ((size_t)(bb * SEQ + kr)) * QKVW + hh * DKH + d8;
  bf16x8 kreg, vreg;
  auto load_kv = [&](int kb) {
    const __bf16* p = kvb + (size_t)kb * QKVW;
    kreg = *(const bf16x8*)(p + DMODEL);       // K
    vreg = *(const bf16x8*)(p + 2 * DMODEL);   // V
  };

  float mi[8], li[8];
  v8f o[4];
#pragma unroll
  for (int r = 0; r < 8; ++r) { mi[r] = -3.0e38f; li[r] = 0.f; }
#pragma unroll
  for (int t = 0; t < 4; ++t)
#pragma unroll
    for (int r = 0; r < 8; ++r) o[t][r] = 0.f;

  load_kv(0);
  for (int kb = 0; kb < SEQ; kb += 32) {
    __syncthreads();
    *(bf16x8*)&Ks[kr][d8] = kreg;              // one b128 store
#pragma unroll
    for (int j = 0; j < 8; ++j) Vt[d8 + j][kr] = vreg[j];  // transposed
    __syncthreads();
    if (kb + 32 < SEQ) load_kv(kb + 32);       // prefetch next block

    // scores: two 16x16 tiles over the 32 k-columns, K-dim 64 = 2 WMMAs each
    float st[2][8];
#pragma unroll
    for (int kt = 0; kt < 2; ++kt) {
      const __bf16* kp = &Ks[kt * 16 + lcol][0];
      v16bf b0 = cat8(*(const bf16x8*)(kp + hl * 8),
                      *(const bf16x8*)(kp + 16 + hl * 8));
      v16bf b1 = cat8(*(const bf16x8*)(kp + 32 + hl * 8),
                      *(const bf16x8*)(kp + 48 + hl * 8));
      v8f sv;
#pragma unroll
      for (int r = 0; r < 8; ++r) sv[r] = 0.f;
      sv = wmma_bf16(a0, b0, sv);
      sv = wmma_bf16(a1, b1, sv);
#pragma unroll
      for (int r = 0; r < 8; ++r) st[kt][r] = sv[r];
    }

    // online softmax (each row lives in one 16-lane half -> xor masks 1..8)
#pragma unroll
    for (int r = 0; r < 8; ++r) {
      float cm = fmaxf(st[0][r], st[1][r]);
#pragma unroll
      for (int m = 8; m >= 1; m >>= 1) cm = fmaxf(cm, __shfl_xor(cm, m, 32));
      float nm = fmaxf(mi[r], cm);
      float alpha = __expf(mi[r] - nm);
      mi[r] = nm;
      st[0][r] = __expf(st[0][r] - nm);
      st[1][r] = __expf(st[1][r] - nm);
      float rs = st[0][r] + st[1][r];
#pragma unroll
      for (int m = 8; m >= 1; m >>= 1) rs += __shfl_xor(rs, m, 32);
      li[r] = li[r] * alpha + rs;
#pragma unroll
      for (int t = 0; t < 4; ++t) o[t][r] *= alpha;
      int prow = r + hl * 8;
      Pb[wave][prow][lcol]      = (__bf16)st[0][r];
      Pb[wave][prow][16 + lcol] = (__bf16)st[1][r];
    }
    __syncthreads();

    // P (A fragment) x V -> O, 4 N-tiles of 16 over dk
    const __bf16* pr = &Pb[wave][lcol][0];
    v16bf ap = cat8(*(const bf16x8*)(pr + hl * 8),
                    *(const bf16x8*)(pr + 16 + hl * 8));
#pragma unroll
    for (int t = 0; t < 4; ++t) {
      v16bf bv = cat8(*(const bf16x8*)&Vt[t * 16 + lcol][hl * 8],
                      *(const bf16x8*)&Vt[t * 16 + lcol][16 + hl * 8]);
      o[t] = wmma_bf16(ap, bv, o[t]);
    }
  }

  // normalize + write
#pragma unroll
  for (int r = 0; r < 8; ++r) {
    float inv = 1.0f / li[r];
    int row = qblk + wave * 16 + r + hl * 8;
    size_t orow = ((size_t)(bb * SEQ + row)) * DMODEL + hh * DKH;
#pragma unroll
    for (int t = 0; t < 4; ++t)
      obuf[orow + t * 16 + lcol] = (__bf16)(o[t][r] * inv);
  }
}

// ---------------------------------------------------------------------------
// Copy CLS-token rows of h (f32) to bf16 [16, 768]
// ---------------------------------------------------------------------------
__global__ __launch_bounds__(256) void cls_gather_kernel(
    const float* __restrict__ h, __bf16* __restrict__ out) {
  int b = blockIdx.x;
  size_t src = (size_t)b * SEQ * DMODEL;
#pragma unroll
  for (int j = 0; j < 3; ++j) {
    int c = threadIdx.x + j * 256;
    out[(size_t)b * DMODEL + c] = (__bf16)h[src + c];
  }
}

// ---------------------------------------------------------------------------
// Host-side orchestration
// ---------------------------------------------------------------------------
static void launch_gemm(const __bf16* A, const float* W, const float* bias,
                        const float* resid, float* outF, __bf16* outB,
                        int M, int N, int K, int act, hipStream_t s) {
  dim3 grid((N + GBN - 1) / GBN, (M + GBM - 1) / GBM);
  if ((M % GBM == 0) && (N % GBN == 0))
    gemm_bf16_kernel<true><<<grid, 256, 0, s>>>(A, W, bias, resid, outF, outB,
                                                M, N, K, act);
  else
    gemm_bf16_kernel<false><<<grid, 256, 0, s>>>(A, W, bias, resid, outF, outB,
                                                 M, N, K, act);
}

extern "C" void kernel_launch(void* const* d_in, const int* in_sizes, int n_in,
                              void* d_out, int out_size, void* d_ws, size_t ws_size,
                              hipStream_t stream) {
  (void)in_sizes; (void)n_in; (void)out_size; (void)ws_size;
  const int*   x        = (const int*)d_in[0];
  const int*   tts      = (const int*)d_in[1];
  const float* tok_emb  = (const float*)d_in[3];
  const float* pos_emb  = (const float*)d_in[4];
  const float* seg_emb  = (const float*)d_in[5];
  const float* Wqkv     = (const float*)d_in[6];
  const float* bqkv     = (const float*)d_in[7];
  const float* Wo       = (const float*)d_in[8];
  const float* bo       = (const float*)d_in[9];
  const float* ln1_g    = (const float*)d_in[10];
  const float* ln1_b    = (const float*)d_in[11];
  const float* W1       = (const float*)d_in[12];
  const float* b1       = (const float*)d_in[13];
  const float* W2       = (const float*)d_in[14];
  const float* b2       = (const float*)d_in[15];
  const float* ln2_g    = (const float*)d_in[16];
  const float* ln2_b    = (const float*)d_in[17];
  const float* pooler_W = (const float*)d_in[18];
  const float* pooler_b = (const float*)d_in[19];
  const float* cls_W1   = (const float*)d_in[20];
  const float* cls_b1   = (const float*)d_in[21];
  const float* cls_W2   = (const float*)d_in[22];
  const float* cls_b2   = (const float*)d_in[23];

  // workspace carve-up
  char* w = (char*)d_ws;
  size_t off = 0;
  float*  h    = (float*)(w + off);  off += (size_t)NTOK * DMODEL * 4;   // 25.2 MB
  __bf16* nbuf = (__bf16*)(w + off); off += (size_t)NTOK * DMODEL * 2;   // 12.6 MB
  __bf16* big  = (__bf16*)(w + off); off += (size_t)NTOK * 3072 * 2;     // 50.3 MB (qkv / ff alias)
  __bf16* obuf = (__bf16*)(w + off); off += (size_t)NTOK * DMODEL * 2;   // 12.6 MB
  __bf16* pin    = (__bf16*)(w + off); off += 16 * DMODEL * 2;
  __bf16* pooled = (__bf16*)(w + off); off += 16 * DMODEL * 2;
  __bf16* hid    = (__bf16*)(w + off); off += 16 * DMODEL * 2;

  // embeddings
  emb_kernel<<<NTOK, 256, 0, stream>>>(x, tts, tok_emb, pos_emb, seg_emb, h);

  for (int i = 0; i < 6; ++i) {
    const float* Wqkv_i = Wqkv + (size_t)i * 3 * DMODEL * DMODEL;
    const float* bqkv_i = bqkv + (size_t)i * QKVW;
    const float* Wo_i   = Wo + (size_t)i * DMODEL * DMODEL;
    const float* bo_i   = bo + (size_t)i * DMODEL;
    const float* W1_i   = W1 + (size_t)i * 3072 * DMODEL;
    const float* b1_i   = b1 + (size_t)i * 3072;
    const float* W2_i   = W2 + (size_t)i * DMODEL * 3072;
    const float* b2_i   = b2 + (size_t)i * DMODEL;

    // pre-LN -> nbuf (bf16)
    ln_kernel<<<NTOK, 256, 0, stream>>>(h, ln1_g + i * DMODEL,
                                        ln1_b + i * DMODEL, nbuf);
    // fused QKV: [8192,768] x [2304,768]^T -> big (bf16)
    launch_gemm(nbuf, Wqkv_i, bqkv_i, nullptr, nullptr, big,
                NTOK, QKVW, DMODEL, 0, stream);
    // flash attention -> obuf (bf16)
    attn_kernel<<<dim3(SEQ / 128, NHEAD, 16), 256, 0, stream>>>(big, obuf);
    // output proj + residual -> h (f32)
    launch_gemm(obuf, Wo_i, bo_i, h, h, nullptr,
                NTOK, DMODEL, DMODEL, 0, stream);
    // pre-LN (FFN) -> nbuf
    ln_kernel<<<NTOK, 256, 0, stream>>>(h, ln2_g + i * DMODEL,
                                        ln2_b + i * DMODEL, nbuf);
    // FFN up + exact GELU -> big (bf16)
    launch_gemm(nbuf, W1_i, b1_i, nullptr, nullptr, big,
                NTOK, 3072, DMODEL, 1, stream);
    // FFN down + residual -> h (f32)
    launch_gemm(big, W2_i, b2_i, h, h, nullptr,
                NTOK, DMODEL, 3072, 0, stream);
  }

  // pooler: tanh(h[:,0,:] @ Wp^T + bp)
  cls_gather_kernel<<<16, 256, 0, stream>>>(h, pin);
  launch_gemm(pin, pooler_W, pooler_b, nullptr, nullptr, pooled,
              16, DMODEL, DMODEL, 2, stream);
  // classifier: ReLU(Linear) -> Linear
  launch_gemm(pooled, cls_W1, cls_b1, nullptr, nullptr, hid,
              16, DMODEL, DMODEL, 3, stream);
  launch_gemm(hid, cls_W2, cls_b2, nullptr, (float*)d_out, nullptr,
              16, 2, DMODEL, 0, stream);
}
